// C2f_DualModal_MoE_59751585022467
// MI455X (gfx1250) — matrix operand order repaired and checked by reading, not compiled, not run on hardware
//
#include <hip/hip_runtime.h>
#include <hip/hip_bf16.h>

typedef __attribute__((ext_vector_type(16))) __bf16 v16bf;
typedef __attribute__((ext_vector_type(8)))  __bf16 v8bf;
typedef __attribute__((ext_vector_type(8)))  float  v8f;

#define BATCH 32
#define C1V   256
#define C2V   256
#define CHV   128
#define NRV   3
#define HGT   80
#define WID   80
#define HWV   6400
#define LDA   40   // A-tile row stride (elements): 80B = 16B-aligned, bank-conflict-free
#define LDB   40   // B-tile (transposed) row stride

// SiLU with hardware reciprocal (v_rcp_f32) instead of IEEE division.
__device__ __forceinline__ float silu_f(float x) {
  return x * __builtin_amdgcn_rcpf(1.0f + __expf(-x));
}

// ---- WMMA fragment loaders (CDNA5 ISA 7.12.2 layouts, wave32) ----
// A tile row-major [128][LDA]; lane reads two contiguous 16B chunks of row m.
template <int LD>
__device__ __forceinline__ v16bf load_a_frag(const __bf16* As, int m_sub, int lane) {
  const int half = lane >> 4, mr = lane & 15;
  const __bf16* row = As + (m_sub + mr) * LD;
  const v8bf lo = *(const v8bf*)(row + half * 8);        // k = half*8 + 0..7
  const v8bf hi = *(const v8bf*)(row + 16 + half * 8);   // k = 16 + half*8 + 0..7
  return __builtin_shufflevector(lo, hi, 0, 1, 2, 3, 4, 5, 6, 7,
                                 8, 9, 10, 11, 12, 13, 14, 15);
}
// B tile TRANSPOSED in LDS: Bt[n][k], [64][LDB]; lane reads k = half*8 + 0..7 and 16 + half*8 + 0..7 of row n.
template <int LD>
__device__ __forceinline__ v16bf load_b_frag(const __bf16* Bt, int n_sub, int lane) {
  const int half = lane >> 4, nc = lane & 15;
  const __bf16* row = Bt + (n_sub + nc) * LD + half * 8;
  const v8bf lo = *(const v8bf*)(row + 0);
  const v8bf hi = *(const v8bf*)(row + 16);
  return __builtin_shufflevector(lo, hi, 0, 1, 2, 3, 4, 5, 6, 7,
                                 8, 9, 10, 11, 12, 13, 14, 15);
}

// ================= Kernel 1: 1x1 conv 256->256 + SiLU, out stored bf16 ==============
__global__ __launch_bounds__(256) void k1_conv1_silu(const float* __restrict__ x,
                                                     const float* __restrict__ w1,
                                                     const float* __restrict__ b1,
                                                     __bf16* __restrict__ t) {
  const int p0 = blockIdx.x * 64;
  const int m0 = blockIdx.y * 128;
  const int b  = blockIdx.z;
  const int tid = threadIdx.x, lane = tid & 31, wave = tid >> 5;

  __shared__ __align__(16) __bf16 As[128 * LDA];
  __shared__ __align__(16) __bf16 Bt[64 * LDB];

  v8f acc[4] = {};
  const float* xb = x + (size_t)b * C1V * HWV;
  const int ar = tid >> 1, ac = (tid & 1) * 16;  // A staging: 16 elems of one row
  const int bn = tid & 63, bkg = tid >> 6;       // B staging: 8 k's of one column

  for (int k0 = 0; k0 < C1V; k0 += 32) {
    if (k0 + 32 < C1V)
      __builtin_prefetch(xb + (size_t)(k0 + 32 + bkg * 8) * HWV + p0 + bn, 0, 1);
    {  // A: w1[m0+ar][k0+ac..+15]  (fp32 -> bf16, two 16B LDS stores)
      v8bf a0, a1;
#pragma unroll
      for (int i = 0; i < 8; ++i) {
        a0[i] = (__bf16)w1[(size_t)(m0 + ar) * C1V + k0 + ac + i];
        a1[i] = (__bf16)w1[(size_t)(m0 + ar) * C1V + k0 + ac + 8 + i];
      }
      *(v8bf*)&As[ar * LDA + ac]     = a0;
      *(v8bf*)&As[ar * LDA + ac + 8] = a1;
    }
    {  // B (transposed): Bt[bn][bkg*8 .. +7] = x[b][k0+bkg*8+i][p0+bn]
      v8bf vv;
#pragma unroll
      for (int i = 0; i < 8; ++i)
        vv[i] = (__bf16)xb[(size_t)(k0 + bkg * 8 + i) * HWV + p0 + bn];
      *(v8bf*)&Bt[bn * LDB + bkg * 8] = vv;
    }
    __syncthreads();
    const v16bf a = load_a_frag<LDA>(As, wave * 16, lane);
    v16bf bb[4];
#pragma unroll
    for (int nt = 0; nt < 4; ++nt) bb[nt] = load_b_frag<LDB>(Bt, nt * 16, lane);
#pragma unroll
    for (int nt = 0; nt < 4; ++nt)
      acc[nt] = __builtin_amdgcn_wmma_f32_16x16x32_bf16(false, a, false, bb[nt], (short)0,
                                                        acc[nt], false, false);
    __syncthreads();
  }
  const int half = lane >> 4, nc = lane & 15;
  __bf16* tb = t + (size_t)b * C1V * HWV;
#pragma unroll
  for (int nt = 0; nt < 4; ++nt)
#pragma unroll
    for (int r = 0; r < 8; ++r) {
      const int m = m0 + wave * 16 + r + half * 8;
      const int n = p0 + nt * 16 + nc;
      tb[(size_t)m * HWV + n] = (__bf16)silu_f(acc[nt][r] + b1[m]);
    }
}

// ================= Kernel 2a: global average pool of feat ==============
__global__ __launch_bounds__(256) void k2a_pool(const __bf16* __restrict__ t,
                                                float* __restrict__ pooled) {
  const int c = blockIdx.x, b = blockIdx.y;
  const __bf16* f = t + (size_t)b * C1V * HWV + (size_t)(CHV + c) * HWV;
  float s = 0.f;
  for (int p = threadIdx.x; p < HWV; p += 256) s += (float)f[p];
  __shared__ float red[256];
  red[threadIdx.x] = s;
  __syncthreads();
  for (int st = 128; st > 0; st >>= 1) {
    if (threadIdx.x < st) red[threadIdx.x] += red[threadIdx.x + st];
    __syncthreads();
  }
  if (threadIdx.x == 0) pooled[b * CHV + c] = red[0] * (1.0f / HWV);
}

// ================= Kernel 2b: router (softmax + top-1) ==============
__global__ void k2b_route(const float* __restrict__ pooled, const float* __restrict__ wr,
                          const float* __restrict__ br, int* __restrict__ ridx,
                          float* __restrict__ rwt) {
  const int b = threadIdx.x;
  if (b >= BATCH) return;
  float l[NRV];
#pragma unroll
  for (int e = 0; e < NRV; ++e) {
    float s = br[e];
    for (int c = 0; c < CHV; ++c) s += pooled[b * CHV + c] * wr[c * NRV + e];
    l[e] = s;
  }
  float mx = fmaxf(l[0], fmaxf(l[1], l[2]));
  float ex[NRV], sum = 0.f;
#pragma unroll
  for (int e = 0; e < NRV; ++e) { ex[e] = __expf(l[e] - mx); sum += ex[e]; }
  int best = 0;
  float bp = ex[0];
#pragma unroll
  for (int e = 1; e < NRV; ++e)
    if (ex[e] > bp) { bp = ex[e]; best = e; }
  ridx[b] = best;
  rwt[b] = bp * __builtin_amdgcn_rcpf(sum);
}

// ====== Kernel 3: shared 3x3 conv + routed-expert 3x3 conv (implicit GEMM, K=1152) ======
__global__ __launch_bounds__(256) void k3_moe_conv3(const __bf16* __restrict__ t,
                                                    const float* __restrict__ wsh,
                                                    const float* __restrict__ bsh,
                                                    const float* __restrict__ we,
                                                    const float* __restrict__ be,
                                                    const int* __restrict__ ridx,
                                                    const float* __restrict__ rwt,
                                                    __bf16* __restrict__ moe) {
  const int p0 = blockIdx.x * 64;
  const int b  = blockIdx.y;
  const int tid = threadIdx.x, lane = tid & 31, wave = tid >> 5;
  const int   e  = ridx[b];
  const float ew = rwt[b];
  const __bf16* feat = t + (size_t)b * C1V * HWV + (size_t)CHV * HWV;
  const float*  wex  = we + (size_t)e * CHV * CHV * 9;
  const int Ktot = CHV * 9;  // 1152

  __shared__ __align__(16) __bf16 Ash[128 * LDA];
  __shared__ __align__(16) __bf16 Aex[128 * LDA];
  __shared__ __align__(16) __bf16 Bt[64 * LDB];

  v8f acc_s[4] = {}, acc_e[4] = {};
  const int ar = tid >> 1, ac = (tid & 1) * 16;
  const int bn = tid & 63, bkg = tid >> 6;
  const int p = p0 + bn, py = p / WID, px = p % WID;

  for (int k0 = 0; k0 < Ktot; k0 += 32) {
    {  // A tiles: shared + selected expert weights
      v8bf s0, s1, e0, e1;
#pragma unroll
      for (int i = 0; i < 8; ++i) {
        s0[i] = (__bf16)wsh[(size_t)ar * Ktot + k0 + ac + i];
        s1[i] = (__bf16)wsh[(size_t)ar * Ktot + k0 + ac + 8 + i];
        e0[i] = (__bf16)wex[(size_t)ar * Ktot + k0 + ac + i];
        e1[i] = (__bf16)wex[(size_t)ar * Ktot + k0 + ac + 8 + i];
      }
      *(v8bf*)&Ash[ar * LDA + ac]     = s0;
      *(v8bf*)&Ash[ar * LDA + ac + 8] = s1;
      *(v8bf*)&Aex[ar * LDA + ac]     = e0;
      *(v8bf*)&Aex[ar * LDA + ac + 8] = e1;
    }
    {  // B (transposed im2col): Bt[bn][bkg*8+i], k -> (ci, ky, kx)
      v8bf vv;
#pragma unroll
      for (int i = 0; i < 8; ++i) {
        const int kg = k0 + bkg * 8 + i;
        const int ci = kg / 9, tap = kg % 9;
        const int yy = py + tap / 3 - 1, xs = px + tap % 3 - 1;
        __bf16 v = (__bf16)0.0f;
        if (yy >= 0 && yy < HGT && xs >= 0 && xs < WID)
          v = feat[(size_t)ci * HWV + yy * WID + xs];
        vv[i] = v;
      }
      *(v8bf*)&Bt[bn * LDB + bkg * 8] = vv;
    }
    __syncthreads();
    const v16bf a_s = load_a_frag<LDA>(Ash, wave * 16, lane);
    const v16bf a_e = load_a_frag<LDA>(Aex, wave * 16, lane);
    v16bf bb[4];
#pragma unroll
    for (int nt = 0; nt < 4; ++nt) bb[nt] = load_b_frag<LDB>(Bt, nt * 16, lane);
#pragma unroll
    for (int nt = 0; nt < 4; ++nt) {
      acc_s[nt] = __builtin_amdgcn_wmma_f32_16x16x32_bf16(false, a_s, false, bb[nt], (short)0,
                                                          acc_s[nt], false, false);
      acc_e[nt] = __builtin_amdgcn_wmma_f32_16x16x32_bf16(false, a_e, false, bb[nt], (short)0,
                                                          acc_e[nt], false, false);
    }
    __syncthreads();
  }
  const int half = lane >> 4, nc = lane & 15;
  __bf16* mo = moe + (size_t)b * CHV * HWV;
#pragma unroll
  for (int nt = 0; nt < 4; ++nt)
#pragma unroll
    for (int r = 0; r < 8; ++r) {
      const int m = wave * 16 + r + half * 8;
      const int n = p0 + nt * 16 + nc;
      const float vs = silu_f(acc_s[nt][r] + bsh[m]);
      const float ve = silu_f(acc_e[nt][r] + be[e * CHV + m]);
      mo[(size_t)m * HWV + n] = (__bf16)(vs + ew * ve);
    }
}

// ================= Kernel 4: 1x1 conv 384->256 + SiLU (fp32 out) ==============
__global__ __launch_bounds__(256) void k4_conv2(const __bf16* __restrict__ t,
                                                const __bf16* __restrict__ moe,
                                                const float* __restrict__ w2,
                                                const float* __restrict__ b2,
                                                float* __restrict__ out) {
  const int p0 = blockIdx.x * 64;
  const int m0 = blockIdx.y * 128;
  const int b  = blockIdx.z;
  const int tid = threadIdx.x, lane = tid & 31, wave = tid >> 5;
  const int Ktot = 3 * CHV;  // 384

  __shared__ __align__(16) __bf16 As[128 * LDA];
  __shared__ __align__(16) __bf16 Bt[64 * LDB];

  v8f acc[4] = {};
  const __bf16* tb = t + (size_t)b * C1V * HWV;
  const __bf16* mb = moe + (size_t)b * CHV * HWV;
  const int ar = tid >> 1, ac = (tid & 1) * 16;
  const int bn = tid & 63, bkg = tid >> 6;

  for (int k0 = 0; k0 < Ktot; k0 += 32) {
    {
      v8bf a0, a1;
#pragma unroll
      for (int i = 0; i < 8; ++i) {
        a0[i] = (__bf16)w2[(size_t)(m0 + ar) * Ktot + k0 + ac + i];
        a1[i] = (__bf16)w2[(size_t)(m0 + ar) * Ktot + k0 + ac + 8 + i];
      }
      *(v8bf*)&As[ar * LDA + ac]     = a0;
      *(v8bf*)&As[ar * LDA + ac + 8] = a1;
    }
    {
      v8bf vv;
#pragma unroll
      for (int i = 0; i < 8; ++i) {
        const int ch = k0 + bkg * 8 + i;
        const __bf16* src = (ch < 2 * CHV) ? (tb + (size_t)ch * HWV)
                                           : (mb + (size_t)(ch - 2 * CHV) * HWV);
        vv[i] = src[p0 + bn];
      }
      *(v8bf*)&Bt[bn * LDB + bkg * 8] = vv;
    }
    __syncthreads();
    const v16bf a = load_a_frag<LDA>(As, wave * 16, lane);
    v16bf bb[4];
#pragma unroll
    for (int nt = 0; nt < 4; ++nt) bb[nt] = load_b_frag<LDB>(Bt, nt * 16, lane);
#pragma unroll
    for (int nt = 0; nt < 4; ++nt)
      acc[nt] = __builtin_amdgcn_wmma_f32_16x16x32_bf16(false, a, false, bb[nt], (short)0,
                                                        acc[nt], false, false);
    __syncthreads();
  }
  const int half = lane >> 4, nc = lane & 15;
  float* ob = out + (size_t)b * C2V * HWV;
#pragma unroll
  for (int nt = 0; nt < 4; ++nt)
#pragma unroll
    for (int r = 0; r < 8; ++r) {
      const int m = m0 + wave * 16 + r + half * 8;
      const int n = p0 + nt * 16 + nc;
      ob[(size_t)m * HWV + n] = silu_f(acc[nt][r] + b2[m]);
    }
}

extern "C" void kernel_launch(void* const* d_in, const int* in_sizes, int n_in,
                              void* d_out, int out_size, void* d_ws, size_t ws_size,
                              hipStream_t stream) {
  const float* x   = (const float*)d_in[0];
  const float* w1  = (const float*)d_in[1];
  const float* b1  = (const float*)d_in[2];
  const float* wr  = (const float*)d_in[3];
  const float* br  = (const float*)d_in[4];
  const float* wsh = (const float*)d_in[5];
  const float* bsh = (const float*)d_in[6];
  const float* we  = (const float*)d_in[7];
  const float* be  = (const float*)d_in[8];
  const float* w2  = (const float*)d_in[9];
  const float* b2  = (const float*)d_in[10];
  float* out = (float*)d_out;

  char* base = (char*)d_ws;
  const size_t t_bytes   = (size_t)BATCH * C1V * HWV * sizeof(__bf16);  // ~100 MB
  const size_t moe_bytes = (size_t)BATCH * CHV * HWV * sizeof(__bf16);  // ~50 MB
  __bf16* t      = (__bf16*)base;
  __bf16* moe    = (__bf16*)(base + t_bytes);
  float*  pooled = (float*)(base + t_bytes + moe_bytes);
  int*    ridx   = (int*)(pooled + BATCH * CHV);
  float*  rwt    = (float*)(ridx + BATCH);

  k1_conv1_silu<<<dim3(HWV / 64, 2, BATCH), 256, 0, stream>>>(x, w1, b1, t);
  k2a_pool<<<dim3(CHV, BATCH), 256, 0, stream>>>(t, pooled);
  k2b_route<<<1, 32, 0, stream>>>(pooled, wr, br, ridx, rwt);
  k3_moe_conv3<<<dim3(HWV / 64, BATCH), 256, 0, stream>>>(t, wsh, bsh, we, be, ridx, rwt, moe);
  k4_conv2<<<dim3(HWV / 64, 2, BATCH), 256, 0, stream>>>(t, moe, w2, b2, out);
}
